// SpaceParticleNet_25503515804302
// MI455X (gfx1250) — compile-verified, matrix operand order checked by reading, not run on hardware
//
#include <hip/hip_runtime.h>
#include <hip/hip_bf16.h>

typedef __attribute__((ext_vector_type(16))) _Float16 v16h;
typedef __attribute__((ext_vector_type(8)))  _Float16 v8h;
typedef __attribute__((ext_vector_type(8)))  float    v8f;

#define NEV   64
#define NPT   256
#define NPTS  (NEV * NPT)
#define BN_EPS 1e-3f

__device__ __forceinline__ v16h make_v16(v8h lo, v8h hi) {
    return __builtin_shufflevector(lo, hi, 0,1,2,3,4,5,6,7,8,9,10,11,12,13,14,15);
}
__device__ __forceinline__ v8f wmma_f16(v16h a, v16h b, v8f c) {
    return __builtin_amdgcn_wmma_f32_16x16x32_f16(false, a, false, b, (short)0, c, false, false);
}

// ---------------------------------------------------------------------------
// Prep: xf36 = [x_cat(6), x(30)] as f16 padded to 64; coords; mask
// ---------------------------------------------------------------------------
__global__ void prep_kernel(const float* __restrict__ x, const float* __restrict__ xcat,
                            _Float16* __restrict__ xf, float* __restrict__ coor,
                            float* __restrict__ maskW)
{
    int p = blockIdx.x * blockDim.x + threadIdx.x;
    if (p >= NPTS) return;
    const float* xr = x + (size_t)p * 30;
    const float* cr = xcat + (size_t)p * 6;
    _Float16* o = xf + (size_t)p * 64;
    for (int i = 0; i < 6;  ++i) o[i]     = (_Float16)cr[i];
    for (int i = 0; i < 30; ++i) o[6 + i] = (_Float16)xr[i];
    for (int i = 36; i < 64; ++i) o[i] = (_Float16)0.f;
    coor[2 * p]     = xr[28];
    coor[2 * p + 1] = xr[29];
    maskW[p] = xr[0];
}

__global__ void norm_kernel(const float* __restrict__ maskW, float* __restrict__ norm)
{
    __shared__ float s[NPT];
    int ev = blockIdx.x, t = threadIdx.x;
    s[t] = maskW[ev * NPT + t];
    __syncthreads();
    for (int off = NPT / 2; off > 0; off >>= 1) {
        if (t < off) s[t] += s[t + off];
        __syncthreads();
    }
    if (t == 0) norm[ev] = s[0];
}

// ---------------------------------------------------------------------------
// Pack f32 weight matrix [K][N] row-major into WMMA B-operand layout:
// out[(nt*KT + kt)*32 + lane][16 halves], K padded to KT*32, N to NT*16.
// ---------------------------------------------------------------------------
__global__ void pack_w_kernel(const float* __restrict__ B, int K, int N, int KT, int NT,
                              _Float16* __restrict__ out)
{
    int idx = blockIdx.x;
    int kt = idx % KT, nt = idx / KT;
    int lane = threadIdx.x & 31;
    int n  = nt * 16 + (lane & 15);
    int hh = lane >> 4;
    v16h v;
    for (int j = 0; j < 8; ++j) {
        int k  = kt * 32 + 8 * hh + j;
        int k2 = k + 16;
        v[j]     = (k  < K && n < N) ? (_Float16)B[(size_t)k  * N + n] : (_Float16)0.f;
        v[8 + j] = (k2 < K && n < N) ? (_Float16)B[(size_t)k2 * N + n] : (_Float16)0.f;
    }
    *(v16h*)(out + ((size_t)(nt * KT + kt) * 32 + lane) * 16) = v;
}

// ---------------------------------------------------------------------------
// Pack per-event activation h[ev][256][CP] (f16, row stride CP) into
// B-operand layout: out[ev][nt][kt(8)][32 lanes][16 halves].
// (rows of B = point index j, cols = channels)
// ---------------------------------------------------------------------------
__global__ void pack_act_kernel(const _Float16* __restrict__ h, int CP, int NT,
                                _Float16* __restrict__ out)
{
    int ev = blockIdx.x, nt = blockIdx.y, kt = blockIdx.z;
    int lane = threadIdx.x & 31;
    int nch = nt * 16 + (lane & 15);
    int hh  = lane >> 4;
    const _Float16* basep = h + (size_t)ev * NPT * CP + nch;
    v16h v;
    for (int j = 0; j < 8; ++j) {
        int k  = kt * 32 + 8 * hh + j;
        int k2 = kt * 32 + 16 + 8 * hh + j;
        v[j]     = basep[(size_t)k  * CP];
        v[8 + j] = basep[(size_t)k2 * CP];
    }
    *(v16h*)(out + (((size_t)(ev * NT + nt) * 8 + kt) * 32 + lane) * 16) = v;
}

// ---------------------------------------------------------------------------
// Generic WMMA GEMM: C[16x16 tile] = A @ B (+bias, relu).
// A f16 row-major [M][Kp] (Kp mult of 32); B pre-packed; grid=(M/16, NT), 32 thr.
// ---------------------------------------------------------------------------
__global__ void gemm_wmma_kernel(const _Float16* __restrict__ A, int Kp, int KT,
                                 const _Float16* __restrict__ Bp,
                                 const float* __restrict__ bias, int Nlog,
                                 _Float16* __restrict__ out16, int s16,
                                 float* __restrict__ out32, int s32, int doRelu)
{
    int mt = blockIdx.x, nt = blockIdx.y;
    int lane = threadIdx.x & 31;
    int half = lane >> 4, nl = lane & 15;
    int row = mt * 16 + nl;
    v8f acc = {};
    const _Float16* arow = A + (size_t)row * Kp + 8 * half;
    const _Float16* bp   = Bp + ((size_t)nt * KT) * 512 + (size_t)lane * 16;
    for (int kt = 0; kt < KT; ++kt) {
        v8h alo = *(const v8h*)(arow);
        v8h ahi = *(const v8h*)(arow + 16);
        v16h a = make_v16(alo, ahi);
        v16h b = *(const v16h*)(bp);
        acc = wmma_f16(a, b, acc);
        arow += 32; bp += 512;
    }
    int ch = nt * 16 + nl;
    if (ch < Nlog) {
        float bv = bias ? bias[ch] : 0.f;
        for (int r = 0; r < 8; ++r) {
            int m = mt * 16 + half * 8 + r;
            float v = acc[r] + bv;
            if (doRelu) v = fmaxf(v, 0.f);
            if (out32) out32[(size_t)m * s32 + ch] = v;
            if (out16) out16[(size_t)m * s16 + ch] = (_Float16)v;
        }
    }
}

// ---------------------------------------------------------------------------
// Edge conv: one block = (event, 16-row m-tile), 128 threads (4 waves).
//   Wm[m][j] = exp(-10*|c_m - c_j|^2) * mask[j]   (16x256 strip, packed A)
//   ss = Wm @ h ; ss = (ss - h)/norm ; rows = [h, ss] ; out = rows @ A + b
//   out *= mask[m]; out = relu(bn(out))
// ---------------------------------------------------------------------------
template<int CIN, int CP, int NOUT, int K2P>
__global__ __launch_bounds__(128) void edge_conv_kernel(
    const _Float16* __restrict__ hin,       // [NPTS][CP]
    const _Float16* __restrict__ packedH,   // [NEV][NT1][8][512]
    const float* __restrict__ cbase, int cstride, int coff,
    const float* __restrict__ maskG, const float* __restrict__ normG,
    const _Float16* __restrict__ Bp2,       // packed [NT2][KT2][512]
    const float* __restrict__ bias2,
    const float* __restrict__ bng, const float* __restrict__ bnb,
    const float* __restrict__ bnmu, const float* __restrict__ bnvar,
    float* __restrict__ outF32, _Float16* __restrict__ outF16)
{
    constexpr int NT1 = CP / 16;
    constexpr int KT2 = K2P / 32;
    constexpr int NT2 = NOUT / 16;

    __shared__ __align__(32) _Float16 wmA[8 * 512];     // packed Wm strip
    __shared__ __align__(32) _Float16 rowsS[16 * K2P];  // [16][K2P] concat rows
    __shared__ float cS[NPT * 2];
    __shared__ float maskS[NPT];

    int ev = blockIdx.x, mt = blockIdx.y;
    int tid = threadIdx.x;
    int base = ev * NPT;

    for (int j = tid; j < NPT; j += 128) {
        maskS[j] = maskG[base + j];
        cS[2 * j]     = cbase[(size_t)(base + j) * cstride + coff];
        cS[2 * j + 1] = cbase[(size_t)(base + j) * cstride + coff + 1];
    }
    {
        v8h z = {};
        for (int i = tid; i < 16 * K2P / 8; i += 128) ((v8h*)rowsS)[i] = z;
    }
    __syncthreads();

    // fill h part of concat rows
    for (int i = tid; i < 16 * CIN; i += 128) {
        int r = i / CIN, ch = i % CIN;
        rowsS[r * K2P + ch] = hin[(size_t)(base + mt * 16 + r) * CP + ch];
    }
    // compute Wm strip in packed-A layout: thread -> (m = tid>>3, kt = tid&7)
    {
        int m  = tid >> 3;
        int kt = tid & 7;
        float cx = cS[2 * (mt * 16 + m)];
        float cy = cS[2 * (mt * 16 + m) + 1];
        for (int rr = 0; rr < 32; ++rr) {
            int j = kt * 32 + rr;
            float dx = cx - cS[2 * j], dy = cy - cS[2 * j + 1];
            float w = __expf(-10.f * (dx * dx + dy * dy)) * maskS[j];
            int hh   = (rr >> 3) & 1;
            int slot = (rr & 7) | ((rr >> 4) << 3);
            wmA[(kt * 32 + 16 * hh + m) * 16 + slot] = (_Float16)w;
        }
    }
    __syncthreads();

    float norm = normG[ev];
    int wave = tid >> 5, lane = tid & 31;
    int half = lane >> 4, nl = lane & 15;

    // GEMM1: ss = Wm_strip(16x256) @ h(256xCP)
    for (int nt = wave; nt < NT1; nt += 4) {
        v8f acc = {};
        const _Float16* bp = packedH + (((size_t)(ev * NT1 + nt) * 8) * 32) * 16
                                     + (size_t)lane * 16;
        for (int kt = 0; kt < 8; ++kt) {
            v16h a = *(const v16h*)(wmA + ((size_t)kt * 32 + lane) * 16);
            v16h b = *(const v16h*)(bp);
            acc = wmma_f16(a, b, acc);
            bp += 512;
        }
        int ch = nt * 16 + nl;
        if (ch < CIN) {
            for (int r = 0; r < 8; ++r) {
                int m = half * 8 + r;
                float hv  = (float)rowsS[m * K2P + ch];
                float ssv = (acc[r] - hv) / norm;
                rowsS[m * K2P + CIN + ch] = (_Float16)ssv;
            }
        }
    }
    __syncthreads();

    // GEMM2: out = rows(16 x K2P) @ A(K2P x NOUT) + bias; mask; bn; relu
    for (int nt = wave; nt < NT2; nt += 4) {
        v8f acc = {};
        const _Float16* bp = Bp2 + (size_t)nt * KT2 * 512 + (size_t)lane * 16;
        for (int kt = 0; kt < KT2; ++kt) {
            const _Float16* ar = rowsS + nl * K2P + kt * 32 + 8 * half;
            v8h alo = *(const v8h*)(ar);
            v8h ahi = *(const v8h*)(ar + 16);
            v16h a = make_v16(alo, ahi);
            v16h b = *(const v16h*)(bp);
            acc = wmma_f16(a, b, acc);
            bp += 512;
        }
        int ch = nt * 16 + nl;
        float s  = bng[ch] * rsqrtf(bnvar[ch] + BN_EPS);
        float bv = bias2[ch], mu = bnmu[ch], bb = bnb[ch];
        for (int r = 0; r < 8; ++r) {
            int m  = half * 8 + r;
            int pt = base + mt * 16 + m;
            float v = (acc[r] + bv) * maskS[mt * 16 + m];
            v = fmaxf((v - mu) * s + bb, 0.f);
            outF32[(size_t)pt * NOUT + ch] = v;
            outF16[(size_t)pt * NOUT + ch] = (_Float16)v;
        }
    }
}

// ---------------------------------------------------------------------------
// Misc small kernels
// ---------------------------------------------------------------------------
__global__ void finish66_kernel(const float* __restrict__ coor, _Float16* __restrict__ h66)
{
    int p = blockIdx.x * blockDim.x + threadIdx.x;
    if (p >= NPTS) return;
    _Float16* r = h66 + (size_t)p * 96;
    r[64] = (_Float16)coor[2 * p];
    r[65] = (_Float16)coor[2 * p + 1];
    for (int i = 66; i < 96; ++i) r[i] = (_Float16)0.f;
}

__global__ void concat130_kernel(const _Float16* __restrict__ h66,
                                 const _Float16* __restrict__ e1,
                                 _Float16* __restrict__ conc)
{
    int p = blockIdx.x * blockDim.x + threadIdx.x;
    if (p >= NPTS) return;
    const _Float16* a = h66 + (size_t)p * 96;
    const _Float16* b = e1  + (size_t)p * 64;
    _Float16* o = conc + (size_t)p * 160;
    for (int i = 0; i < 66; ++i) o[i] = a[i];
    for (int i = 0; i < 64; ++i) o[66 + i] = b[i];
    for (int i = 130; i < 160; ++i) o[i] = (_Float16)0.f;
}

__global__ void pool_kernel(const float* __restrict__ h, const float* __restrict__ maskW,
                            float* __restrict__ z)
{
    int ev = blockIdx.x, ch = threadIdx.x;  // 128 threads
    float s = 0.f, mx = -1e30f;
    for (int j = 0; j < NPT; ++j) {
        float v = h[(size_t)(ev * NPT + j) * 128 + ch] * maskW[ev * NPT + j];
        s += v; mx = fmaxf(mx, v);
    }
    z[ev * 256 + ch]       = s * (1.f / (float)NPT);
    z[ev * 256 + 128 + ch] = mx;
}

__global__ void f2h_kernel(const float* __restrict__ in, _Float16* __restrict__ out, int n)
{
    int i = blockIdx.x * blockDim.x + threadIdx.x;
    if (i < n) out[i] = (_Float16)in[i];
}

__global__ void bn_relu_kernel(const float* __restrict__ in, int C,
                               const float* __restrict__ g, const float* __restrict__ b,
                               const float* __restrict__ mu, const float* __restrict__ var,
                               _Float16* __restrict__ out16, int n)
{
    int i = blockIdx.x * blockDim.x + threadIdx.x;
    if (i >= n) return;
    int ch = i % C;
    float s = g[ch] * rsqrtf(var[ch] + BN_EPS);
    float v = fmaxf((in[i] - mu[ch]) * s + b[ch], 0.f);
    out16[i] = (_Float16)v;
}

__global__ void softmax_kernel(const float* __restrict__ logits, float* __restrict__ out)
{
    int r = blockIdx.x * blockDim.x + threadIdx.x;
    if (r >= NEV) return;
    float v[4]; float mx = -1e30f;
    for (int i = 0; i < 4; ++i) { v[i] = logits[r * 16 + i]; mx = fmaxf(mx, v[i]); }
    float s = 0.f;
    for (int i = 0; i < 4; ++i) { v[i] = __expf(v[i] - mx); s += v[i]; }
    float inv = 1.f / s;
    for (int i = 0; i < 4; ++i) out[r * 4 + i] = v[i] * inv;
}

// ---------------------------------------------------------------------------
extern "C" void kernel_launch(void* const* d_in, const int* in_sizes, int n_in,
                              void* d_out, int out_size, void* d_ws, size_t ws_size,
                              hipStream_t stream)
{
    (void)in_sizes; (void)n_in; (void)out_size; (void)ws_size;
    const float* x    = (const float*)d_in[0];
    const float* xcat = (const float*)d_in[1];
    const float* W1 = (const float*)d_in[2];  const float* b1 = (const float*)d_in[3];
    const float* W2 = (const float*)d_in[4];  const float* b2 = (const float*)d_in[5];
    const float* A0 = (const float*)d_in[6];  const float* ab0 = (const float*)d_in[7];
    const float* A1 = (const float*)d_in[8];  const float* ab1 = (const float*)d_in[9];
    const float* A2 = (const float*)d_in[10]; const float* ab2 = (const float*)d_in[11];
    const float *bn0g=(const float*)d_in[12], *bn0b=(const float*)d_in[13],
                *bn0m=(const float*)d_in[14], *bn0v=(const float*)d_in[15];
    const float *bn1g=(const float*)d_in[16], *bn1b=(const float*)d_in[17],
                *bn1m=(const float*)d_in[18], *bn1v=(const float*)d_in[19];
    const float *bn2g=(const float*)d_in[20], *bn2b=(const float*)d_in[21],
                *bn2m=(const float*)d_in[22], *bn2v=(const float*)d_in[23];
    const float* D1 = (const float*)d_in[24]; const float* d1b = (const float*)d_in[25];
    const float* D2 = (const float*)d_in[26]; const float* d2b = (const float*)d_in[27];
    const float *dn0g=(const float*)d_in[28], *dn0b=(const float*)d_in[29],
                *dn0m=(const float*)d_in[30], *dn0v=(const float*)d_in[31];
    const float *dn1g=(const float*)d_in[32], *dn1b=(const float*)d_in[33],
                *dn1m=(const float*)d_in[34], *dn1v=(const float*)d_in[35];
    const float* OW = (const float*)d_in[36]; const float* OB = (const float*)d_in[37];

    char* ws = (char*)d_ws; size_t off = 0;
    auto alloc = [&](size_t bytes) -> void* {
        void* p = ws + off; off = (off + bytes + 255) & ~(size_t)255; return p;
    };

    _Float16* xf36h = (_Float16*)alloc((size_t)NPTS * 64 * 2);
    float*    coor  = (float*)   alloc((size_t)NPTS * 2 * 4);
    float*    maskW = (float*)   alloc((size_t)NPTS * 4);
    float*    normW = (float*)   alloc((size_t)NEV * 4);
    _Float16* h1h   = (_Float16*)alloc((size_t)NPTS * 64 * 2);
    _Float16* h66h  = (_Float16*)alloc((size_t)NPTS * 96 * 2);
    float*    e0f   = (float*)   alloc((size_t)NPTS * 64 * 4);
    _Float16* e0h   = (_Float16*)alloc((size_t)NPTS * 64 * 2);
    float*    e1f   = (float*)   alloc((size_t)NPTS * 64 * 4);
    _Float16* e1h   = (_Float16*)alloc((size_t)NPTS * 64 * 2);
    _Float16* conc  = (_Float16*)alloc((size_t)NPTS * 160 * 2);
    float*    e2f   = (float*)   alloc((size_t)NPTS * 128 * 4);
    _Float16* e2h   = (_Float16*)alloc((size_t)NPTS * 128 * 2);
    _Float16* packedH = (_Float16*)alloc((size_t)NEV * 10 * 8 * 512 * 2); // max NT1=10
    _Float16* W1p = (_Float16*)alloc((size_t)2 * 4  * 512 * 2);
    _Float16* W2p = (_Float16*)alloc((size_t)2 * 4  * 512 * 2);
    _Float16* A0p = (_Float16*)alloc((size_t)5 * 4  * 512 * 2);
    _Float16* A1p = (_Float16*)alloc((size_t)4 * 4  * 512 * 2);
    _Float16* A2p = (_Float16*)alloc((size_t)9 * 8  * 512 * 2);
    _Float16* D1p = (_Float16*)alloc((size_t)8 * 16 * 512 * 2);
    _Float16* D2p = (_Float16*)alloc((size_t)8 * 8  * 512 * 2);
    _Float16* OWp = (_Float16*)alloc((size_t)4 * 1  * 512 * 2);
    float*    zf   = (float*)   alloc((size_t)NEV * 256 * 4);
    _Float16* zh   = (_Float16*)alloc((size_t)NEV * 256 * 2);
    float*    d1f  = (float*)   alloc((size_t)NEV * 256 * 4);
    _Float16* d1h  = (_Float16*)alloc((size_t)NEV * 256 * 2);
    float*    d2f  = (float*)   alloc((size_t)NEV * 128 * 4);
    _Float16* d2h  = (_Float16*)alloc((size_t)NEV * 128 * 2);
    float*    logits = (float*) alloc((size_t)NEV * 16 * 4);

    // prep + mask norms
    prep_kernel<<<NPTS / 256, 256, 0, stream>>>(x, xcat, xf36h, coor, maskW);
    norm_kernel<<<NEV, NPT, 0, stream>>>(maskW, normW);

    // pack weights (K, N, KT, NT)
    pack_w_kernel<<<2 * 4, 32, 0, stream>>>(W1, 36, 64, 2, 4, W1p);
    pack_w_kernel<<<2 * 4, 32, 0, stream>>>(W2, 64, 64, 2, 4, W2p);
    pack_w_kernel<<<5 * 4, 32, 0, stream>>>(A0, 132, 64, 5, 4, A0p);
    pack_w_kernel<<<4 * 4, 32, 0, stream>>>(A1, 128, 64, 4, 4, A1p);
    pack_w_kernel<<<9 * 8, 32, 0, stream>>>(A2, 260, 128, 9, 8, A2p);
    pack_w_kernel<<<8 * 16, 32, 0, stream>>>(D1, 256, 256, 8, 16, D1p);
    pack_w_kernel<<<8 * 8, 32, 0, stream>>>(D2, 256, 128, 8, 8, D2p);
    pack_w_kernel<<<4 * 1, 32, 0, stream>>>(OW, 128, 4, 4, 1, OWp);

    // conv1d MLP (pointwise): 36->64 relu, 64->64 relu, then append coords
    gemm_wmma_kernel<<<dim3(NPTS / 16, 4), 32, 0, stream>>>(
        xf36h, 64, 2, W1p, b1, 64, h1h, 64, nullptr, 0, 1);
    gemm_wmma_kernel<<<dim3(NPTS / 16, 4), 32, 0, stream>>>(
        h1h, 64, 2, W2p, b2, 64, h66h, 96, nullptr, 0, 1);
    finish66_kernel<<<NPTS / 256, 256, 0, stream>>>(coor, h66h);

    // edge conv layer 0: CIN=66, CP=96, NOUT=64, K2P=160
    pack_act_kernel<<<dim3(NEV, 6, 8), 32, 0, stream>>>(h66h, 96, 6, packedH);
    edge_conv_kernel<66, 96, 64, 160><<<dim3(NEV, 16), 128, 0, stream>>>(
        h66h, packedH, coor, 2, 0, maskW, normW,
        A0p, ab0, bn0g, bn0b, bn0m, bn0v, e0f, e0h);

    // edge conv layer 1: CIN=64, CP=64, NOUT=64, K2P=128; coords = ch 62,63 of e0
    pack_act_kernel<<<dim3(NEV, 4, 8), 32, 0, stream>>>(e0h, 64, 4, packedH);
    edge_conv_kernel<64, 64, 64, 128><<<dim3(NEV, 16), 128, 0, stream>>>(
        e0h, packedH, e0f, 64, 62, maskW, normW,
        A1p, ab1, bn1g, bn1b, bn1m, bn1v, e1f, e1h);

    // edge conv layer 2: input = [x0(66), h(64)] = 130 ch; coords = ch 62,63 of e1
    concat130_kernel<<<NPTS / 256, 256, 0, stream>>>(h66h, e1h, conc);
    pack_act_kernel<<<dim3(NEV, 10, 8), 32, 0, stream>>>(conc, 160, 10, packedH);
    edge_conv_kernel<130, 160, 128, 288><<<dim3(NEV, 16), 128, 0, stream>>>(
        conc, packedH, e1f, 64, 62, maskW, normW,
        A2p, ab2, bn2g, bn2b, bn2m, bn2v, e2f, e2h);

    // pooling: mean + max over points -> z[64][256]
    pool_kernel<<<NEV, 128, 0, stream>>>(e2f, maskW, zf);
    f2h_kernel<<<(NEV * 256 + 255) / 256, 256, 0, stream>>>(zf, zh, NEV * 256);

    // dense 256->256 (bn relu), 256->128 (bn relu), head 128->4 softmax
    gemm_wmma_kernel<<<dim3(NEV / 16, 16), 32, 0, stream>>>(
        zh, 256, 8, D1p, d1b, 256, nullptr, 0, d1f, 256, 0);
    bn_relu_kernel<<<(NEV * 256 + 255) / 256, 256, 0, stream>>>(
        d1f, 256, dn0g, dn0b, dn0m, dn0v, d1h, NEV * 256);
    gemm_wmma_kernel<<<dim3(NEV / 16, 8), 32, 0, stream>>>(
        d1h, 256, 8, D2p, d2b, 128, nullptr, 0, d2f, 128, 0);
    bn_relu_kernel<<<(NEV * 128 + 255) / 256, 256, 0, stream>>>(
        d2f, 128, dn1g, dn1b, dn1m, dn1v, d2h, NEV * 128);
    gemm_wmma_kernel<<<dim3(NEV / 16, 1), 32, 0, stream>>>(
        d2h, 128, 4, OWp, OB, 4, nullptr, 0, logits, 16, 0);
    softmax_kernel<<<1, NEV, 0, stream>>>(logits, (float*)d_out);
}
